// SeqAttnMatch_11175504904492
// MI455X (gfx1250) — matrix operand order repaired and checked by reading, not compiled, not run on hardware
//
#include <hip/hip_runtime.h>
#include <hip/hip_bf16.h>
#include <hip/hip_fp16.h>

// ---------------------------------------------------------------------------
// SeqAttnMatch fused kernels for MI455X (gfx1250, wave32, WMMA f16->f32)
//   B=16, L1=L2=2048, H=128
//   K1: p = relu(z @ W^T + b) -> f16 row-major (x_proj, y_proj in ws)
//   K2: yt[b][h][j] = (f16) y[b][j][h]   (LDS-staged transpose)
//   K3: flash attention: S = x_proj y_proj^T, masked online softmax,
//       O = P @ y (f32 accumulate via f16 WMMA), out = O / l
// ---------------------------------------------------------------------------

typedef _Float16 v16h __attribute__((ext_vector_type(16)));
typedef _Float16 v8h  __attribute__((ext_vector_type(8)));
typedef float    v8f  __attribute__((ext_vector_type(8)));
typedef float    v4f  __attribute__((ext_vector_type(4)));

static constexpr int kB  = 16;
static constexpr int kL1 = 2048;
static constexpr int kL2 = 2048;
static constexpr int kH  = 128;

#define NEG_INF (-__builtin_inff())

// WMMA f16 16x16x32 VGPR layouts (ISA 7.12.2), lane = g*16 + (m|n):
//   A: halves 0..7 -> k = 8g+h ; halves 8..15 -> k = 16+8g+(h-8)
//      (two contiguous 16B chunks per lane from a row-major K source)
//   B: same k order as A; K3 uses k = 16g+h for both A and B (one contiguous 32B chunk)
//   C/D f32: v8f slot r holds row M = r + 8g, column N = n.

// ---------------------------------------------------------------------------
// Kernel 1: shared linear + ReLU projection of x and y, f32 in -> f16 out
// ---------------------------------------------------------------------------
__global__ __launch_bounds__(128) void seqattn_proj_kernel(
    const float* __restrict__ x, const float* __restrict__ y,
    const float* __restrict__ W, const float* __restrict__ bias,
    _Float16* __restrict__ xp, _Float16* __restrict__ yp)
{
    const int lane = threadIdx.x & 31;
    const int wave = threadIdx.x >> 5;
    const int g    = lane >> 4;
    const int ln   = lane & 15;

    const int tile = blockIdx.x * 4 + wave;   // 16-row tile of concat(x,y)
    const int row0 = tile * 16;
    const int BL1  = kB * kL1;

    const float* src;
    _Float16*    dst;
    int          lrow0;
    if (row0 < BL1) { src = x; dst = xp; lrow0 = row0; }
    else            { src = y; dst = yp; lrow0 = row0 - BL1; }

    // ---- A tiles (16 rows x K=128, f32 -> f16), register-resident ----
    v16h a[4];
    {
        const float* pr = src + (size_t)(lrow0 + ln) * kH;
#pragma unroll
        for (int kt = 0; kt < 4; ++kt) {
            const float* p0 = pr + kt * 32 + 8 * g;
            v4f c0 = *(const v4f*)(p0);
            v4f c1 = *(const v4f*)(p0 + 4);
            v4f c2 = *(const v4f*)(p0 + 16);
            v4f c3 = *(const v4f*)(p0 + 20);
            v16h av;
#pragma unroll
            for (int e = 0; e < 4; ++e) {
                av[e]      = (_Float16)c0[e];
                av[4 + e]  = (_Float16)c1[e];
                av[8 + e]  = (_Float16)c2[e];
                av[12 + e] = (_Float16)c3[e];
            }
            a[kt] = av;
        }
    }

    // ---- out[i][o] = relu(sum_h z[i][h] * W[o][h] + b[o]) ----
#pragma unroll
    for (int nt = 0; nt < 8; ++nt) {
        const int ocol = nt * 16 + ln;
        v8f acc = {};
#pragma unroll
        for (int kt = 0; kt < 4; ++kt) {
            const float* wr = W + (size_t)ocol * kH + kt * 32 + 8 * g;
            v4f w0 = *(const v4f*)(wr);
            v4f w1 = *(const v4f*)(wr + 4);
            v4f w2 = *(const v4f*)(wr + 16);
            v4f w3 = *(const v4f*)(wr + 20);
            v16h bv;
#pragma unroll
            for (int e = 0; e < 4; ++e) {
                bv[e]      = (_Float16)w0[e];
                bv[4 + e]  = (_Float16)w1[e];
                bv[8 + e]  = (_Float16)w2[e];
                bv[12 + e] = (_Float16)w3[e];
            }
            acc = __builtin_amdgcn_wmma_f32_16x16x32_f16(
                false, a[kt], false, bv, (short)0, acc, false, false);
        }
        const float bb = bias[ocol];
#pragma unroll
        for (int r = 0; r < 8; ++r) {
            float v = acc[r] + bb;
            v = v > 0.f ? v : 0.f;
            dst[(size_t)(lrow0 + r + 8 * g) * kH + ocol] = (_Float16)v;
        }
    }
}

// ---------------------------------------------------------------------------
// Kernel 2: yt[b][h][j] = (f16) y[b][j][h].  64-j x 128-h tile per block.
// LDS row pitch 130 halves = 65 dwords -> bank stride 1 for j-major reads.
// ---------------------------------------------------------------------------
__global__ __launch_bounds__(256) void seqattn_ytrans_kernel(
    const float* __restrict__ y, _Float16* __restrict__ yt)
{
    __shared__ _Float16 lds[64 * 130];
    const int t  = threadIdx.x;
    const int b  = blockIdx.x >> 5;           // 32 j-tiles per batch
    const int j0 = (blockIdx.x & 31) * 64;

    const float* yb = y + ((size_t)b * kL2 + j0) * kH;
    // Phase 1: coalesced f32 reads (consecutive t -> consecutive h)
#pragma unroll
    for (int r = 0; r < 32; ++r) {
        const int idx = r * 256 + t;          // 0 .. 8191 over 64x128 tile
        const int j   = idx >> 7;
        const int h   = idx & 127;
        lds[j * 130 + h] = (_Float16)yb[(size_t)j * kH + h];
    }
    __syncthreads();
    // Phase 2: coalesced f16 writes (consecutive t -> consecutive j)
    _Float16* ytb = yt + (size_t)b * kH * kL2 + j0;
#pragma unroll
    for (int r = 0; r < 32; ++r) {
        const int idx = r * 256 + t;          // 0 .. 8191 over 128x64 tile
        const int h   = idx >> 6;
        const int j   = idx & 63;
        ytb[(size_t)h * kL2 + j] = lds[j * 130 + h];
    }
}

// ---------------------------------------------------------------------------
// Kernel 3: fused masked attention (flash style), one wave per 16-row i-tile
// ---------------------------------------------------------------------------
__global__ __launch_bounds__(128) void seqattn_attn_kernel(
    const _Float16* __restrict__ xp, const _Float16* __restrict__ yp,
    const _Float16* __restrict__ yt, const unsigned char* __restrict__ y_mask,
    float* __restrict__ out)
{
    __shared__ _Float16 lds_p[4][16 * 32];   // per-wave P transpose slab (1KB)

    const int lane = threadIdx.x & 31;
    const int wave = threadIdx.x >> 5;
    const int g    = lane >> 4;
    const int ln   = lane & 15;

    const int itile = blockIdx.x * 4 + wave;       // 0 .. B*L1/16-1
    const int b     = itile / (kL1 / 16);
    const int i0    = (itile % (kL1 / 16)) * 16;

    const _Float16* xpb = xp + (size_t)b * kL1 * kH;
    const _Float16* ypb = yp + (size_t)b * kL2 * kH;
    const _Float16* ytb = yt + (size_t)b * kH * kL2;
    const unsigned char* mb = y_mask + (size_t)b * kL2;

    // ---- register-resident A tiles of x_proj (16 x 128) ----
    v16h axp[4];
    {
        const _Float16* pr = xpb + (size_t)(i0 + ln) * kH;
#pragma unroll
        for (int kt = 0; kt < 4; ++kt) {
            v8h c0 = *(const v8h*)(pr + kt * 32 + 16 * g);
            v8h c1 = *(const v8h*)(pr + kt * 32 + 16 * g + 8);
            v16h av;
#pragma unroll
            for (int e = 0; e < 8; ++e) { av[e] = c0[e]; av[8 + e] = c1[e]; }
            axp[kt] = av;
        }
    }

    v8f acc[8];
#pragma unroll
    for (int ot = 0; ot < 8; ++ot) acc[ot] = (v8f){};
    float mrow[8], lrow[8];
#pragma unroll
    for (int r = 0; r < 8; ++r) { mrow[r] = NEG_INF; lrow[r] = 0.f; }

    _Float16* myP = lds_p[wave];

    for (int j0 = 0; j0 < kL2; j0 += 32) {
        if (j0 + 32 < kL2) {   // pull next chunk toward L0 (it is L2-resident)
            __builtin_prefetch(ypb + (size_t)(j0 + 32 + ln) * kH, 0, 3);
            __builtin_prefetch(ytb + (size_t)(16 * g + ln) * kL2 + j0 + 32, 0, 3);
        }

        // ---- S tile 16x32: two 16x16 f32 tiles, K=128 over h ----
        v8f s[2];
#pragma unroll
        for (int t = 0; t < 2; ++t) {
            v8f sc = (v8f){};
            const _Float16* br = ypb + (size_t)(j0 + t * 16 + ln) * kH + 16 * g;
#pragma unroll
            for (int kt = 0; kt < 4; ++kt) {
                v16h bv = *(const v16h*)(br + kt * 32);
                sc = __builtin_amdgcn_wmma_f32_16x16x32_f16(
                    false, axp[kt], false, bv, (short)0, sc, false, false);
            }
            s[t] = sc;
        }

        // ---- mask padded j columns (column n lives in lane ln) ----
        bool msk[2];
        msk[0] = mb[j0 + ln] != 0;
        msk[1] = mb[j0 + 16 + ln] != 0;
#pragma unroll
        for (int t = 0; t < 2; ++t) {
            if (msk[t]) {
#pragma unroll
                for (int r = 0; r < 8; ++r) s[t][r] = NEG_INF;
            }
        }

        // ---- online softmax: row max (reduce across 16 lanes in-group) ----
        float mnew[8], scl[8];
#pragma unroll
        for (int r = 0; r < 8; ++r) {
            float v = fmaxf(s[0][r], s[1][r]);
#pragma unroll
            for (int off = 8; off >= 1; off >>= 1)
                v = fmaxf(v, __shfl_xor(v, off, 32));
            mnew[r] = fmaxf(mrow[r], v);
            // ==-compare keeps all-(-inf) rows NaN-free (scale 1, acc/l stay 0)
            scl[r]  = (mrow[r] == mnew[r]) ? 1.f : __expf(mrow[r] - mnew[r]);
            mrow[r] = mnew[r];
        }

        // ---- P = exp(S - m), masked cols forced to 0; row sums ----
#pragma unroll
        for (int r = 0; r < 8; ++r) {
            float p0 = msk[0] ? 0.f : __expf(s[0][r] - mnew[r]);
            float p1 = msk[1] ? 0.f : __expf(s[1][r] - mnew[r]);
            s[0][r] = p0;
            s[1][r] = p1;
            float v = p0 + p1;
#pragma unroll
            for (int off = 8; off >= 1; off >>= 1)
                v += __shfl_xor(v, off, 32);
            lrow[r] = lrow[r] * scl[r] + v;
        }

        // ---- rescale accumulator ----
#pragma unroll
        for (int ot = 0; ot < 8; ++ot)
#pragma unroll
            for (int r = 0; r < 8; ++r) acc[ot][r] *= scl[r];

        // ---- transpose P (D layout -> A layout) through per-wave LDS ----
#pragma unroll
        for (int t = 0; t < 2; ++t)
#pragma unroll
            for (int r = 0; r < 8; ++r)
                myP[(r + 8 * g) * 32 + t * 16 + ln] = (_Float16)s[t][r];
        asm volatile("s_wait_dscnt 0" ::: "memory");
        v16h ap;
        {
            v8h c0 = *(const v8h*)(myP + ln * 32 + 16 * g);
            v8h c1 = *(const v8h*)(myP + ln * 32 + 16 * g + 8);
#pragma unroll
            for (int e = 0; e < 8; ++e) { ap[e] = c0[e]; ap[8 + e] = c1[e]; }
        }

        // ---- O += P @ y : B[k=j][n=h] = yt[h][j0+k], contiguous 32B/lane ----
#pragma unroll
        for (int ot = 0; ot < 8; ++ot) {
            v16h bv = *(const v16h*)(ytb + (size_t)(ot * 16 + ln) * kL2 +
                                     j0 + 16 * g);
            acc[ot] = __builtin_amdgcn_wmma_f32_16x16x32_f16(
                false, ap, false, bv, (short)0, acc[ot], false, false);
        }
    }

    // ---- out = acc / l ----
    float* ob = out + ((size_t)b * kL1 + i0) * kH;
#pragma unroll
    for (int ot = 0; ot < 8; ++ot) {
#pragma unroll
        for (int r = 0; r < 8; ++r) {
            ob[(size_t)(r + 8 * g) * kH + ot * 16 + ln] = acc[ot][r] / lrow[r];
        }
    }
}

// ---------------------------------------------------------------------------
extern "C" void kernel_launch(void* const* d_in, const int* in_sizes, int n_in,
                              void* d_out, int out_size, void* d_ws, size_t ws_size,
                              hipStream_t stream) {
    const float*         x      = (const float*)d_in[0];
    const float*         y      = (const float*)d_in[1];
    const unsigned char* y_mask = (const unsigned char*)d_in[2]; // numpy bool, 1B
    const float*         W      = (const float*)d_in[3];
    const float*         bias   = (const float*)d_in[4];
    float*               out    = (float*)d_out;

    _Float16* xp = (_Float16*)d_ws;                       // B*L1*H f16 (8.4MB)
    _Float16* yp = xp + (size_t)kB * kL1 * kH;            // B*L2*H f16 (8.4MB)
    _Float16* yt = yp + (size_t)kB * kL2 * kH;            // B*H*L2 f16 (8.4MB)

    // 1) shared projection of x and y: 4096 row-tiles, 4 waves/block
    seqattn_proj_kernel<<<(kB * (kL1 + kL2) / 16) / 4, 128, 0, stream>>>(
        x, y, W, bias, xp, yp);

    // 2) y -> yt transpose (f16): 64-j x 128-h tiles
    seqattn_ytrans_kernel<<<kB * (kL2 / 64), 256, 0, stream>>>(y, yt);

    // 3) fused masked attention: B*L1/16 = 2048 i-tiles, 4 waves/block
    seqattn_attn_kernel<<<(kB * kL1 / 16) / 4, 128, 0, stream>>>(
        xp, yp, yt, y_mask, out);
}
